// Transformer_51788715655679
// MI455X (gfx1250) — matrix operand order repaired and checked by reading, not compiled, not run on hardware
//
#include <hip/hip_runtime.h>
#include <hip/hip_bf16.h>
#include <cstdint>

#define N_  8
#define CT_ 1024
#define L_  1024
#define H_  16
#define G_  8

typedef __attribute__((ext_vector_type(16))) __bf16 v16bf;
typedef __attribute__((ext_vector_type(8)))  float  v8f;
typedef __attribute__((ext_vector_type(4)))  int    vi4;
typedef __attribute__((address_space(1))) vi4 gvi4;   // global 16B chunk
typedef __attribute__((address_space(3))) vi4 lvi4;   // LDS 16B chunk

#if __has_builtin(__builtin_amdgcn_global_load_async_to_lds_b128)
#define HAS_ASYNC_LDS 1
#else
#define HAS_ASYNC_LDS 0
#endif

union BF16x16 {
  v16bf v;
  unsigned short u[16];
  uint4 q[2];
};

__device__ __forceinline__ unsigned short f2bf(float f) {
  unsigned int u = __float_as_uint(f);
  u += 0x7fffu + ((u >> 16) & 1u);   // round-to-nearest-even
  return (unsigned short)(u >> 16);
}

// A-operand (16x32 bf16): lane m = l&15; element j -> K=(l>>4)*16+j (contiguous, as in load_b)
__device__ __forceinline__ v16bf load_a(const unsigned short* row, int off0, int off1) {
  BF16x16 t;
  t.q[0] = *reinterpret_cast<const uint4*>(row + off0);
  t.q[1] = *reinterpret_cast<const uint4*>(row + off1);
  return t.v;
}

// B-operand (32x16 bf16): lane n = l&15 holds K = (l>>4)*16 .. +15 contiguous -> one 32B read.
__device__ __forceinline__ v16bf load_b(const unsigned short* p) {
  BF16x16 t;
  t.q[0] = *reinterpret_cast<const uint4*>(p);
  t.q[1] = *reinterpret_cast<const uint4*>(p + 8);
  return t.v;
}

__device__ __forceinline__ v8f wmma_bf16(v16bf a, v16bf b, v8f c) {
  return __builtin_amdgcn_wmma_f32_16x16x32_bf16(false, a, false, b, (short)0, c,
                                                 false, false);
}

__device__ __forceinline__ void store8bf(unsigned short* p, const float* vals) {
  uint4 w;
  w.x = (unsigned)f2bf(vals[0]) | ((unsigned)f2bf(vals[1]) << 16);
  w.y = (unsigned)f2bf(vals[2]) | ((unsigned)f2bf(vals[3]) << 16);
  w.z = (unsigned)f2bf(vals[4]) | ((unsigned)f2bf(vals[5]) << 16);
  w.w = (unsigned)f2bf(vals[6]) | ((unsigned)f2bf(vals[7]) << 16);
  *reinterpret_cast<uint4*>(p) = w;
}

// 16B global -> LDS copy: gfx1250 async path (GLOBAL_LOAD_ASYNC_TO_LDS_B128).
__device__ __forceinline__ void async_copy16(const void* g, void* l) {
#if HAS_ASYNC_LDS
  __builtin_amdgcn_global_load_async_to_lds_b128((gvi4*)g, (lvi4*)l, 0, 0);
#else
  *reinterpret_cast<uint4*>(l) = *reinterpret_cast<const uint4*>(g);
#endif
}

#if HAS_ASYNC_LDS
#if __has_builtin(__builtin_amdgcn_s_wait_asynccnt)
#define ASYNC_WAIT(n) __builtin_amdgcn_s_wait_asynccnt(n)
#else
#define ASYNC_WAIT(n) asm volatile("s_wait_asynccnt " #n ::: "memory")
#endif
#else
#define ASYNC_WAIT(n) ((void)0)
#endif

// ---------------- BN folding: w' = w*inv0*extra ; bias' = ((b-mean)*inv0+beta)*extra
__global__ void fold_bn_kernel(const float* __restrict__ w, const float* __restrict__ b,
                               const float* __restrict__ gamma, const float* __restrict__ beta,
                               const float* __restrict__ mean, const float* __restrict__ var,
                               int cout, int cinpg, float extra,
                               unsigned short* __restrict__ w_bf, float* __restrict__ bias) {
  const int total = cout * cinpg;
  for (int i = blockIdx.x * blockDim.x + threadIdx.x; i < total;
       i += gridDim.x * blockDim.x) {
    const int co = i / cinpg;
    const float inv0 = gamma[co] * rsqrtf(var[co] + 1e-5f);
    w_bf[i] = f2bf(w[i] * inv0 * extra);
    if ((i % cinpg) == 0)
      bias[co] = ((b[co] - mean[co]) * inv0 + beta[co]) * extra;
  }
}

// ---------------- tokens fp32 [n,c,l] -> bf16 transposed [n,l,c] (LDS-tiled)
__global__ __launch_bounds__(256) void transpose_kernel(const float* __restrict__ tok,
                                                        unsigned short* __restrict__ tok_t) {
  __shared__ float tile[32][33];
  const int n = blockIdx.z;
  const int c0 = blockIdx.y * 32;
  const int l0 = blockIdx.x * 32;
  const int tx = threadIdx.x, ty = threadIdx.y;
  #pragma unroll
  for (int i = ty; i < 32; i += 8)
    tile[i][tx] = tok[((size_t)n * CT_ + c0 + i) * L_ + l0 + tx];
  __syncthreads();
  #pragma unroll
  for (int i = ty; i < 32; i += 8)
    tok_t[((size_t)n * L_ + l0 + i) * CT_ + c0 + tx] = f2bf(tile[tx][i]);
}

// ---------------- QKV grouped conv: per wave one 16(cout) x 16(l) tile, K=128 (4 WMMAs)
__global__ __launch_bounds__(256) void qkv_kernel(
    const unsigned short* __restrict__ tok_t,
    const unsigned short* __restrict__ kw, const float* __restrict__ kb,
    const unsigned short* __restrict__ qw, const float* __restrict__ qb,
    const unsigned short* __restrict__ vw, const float* __restrict__ vb,
    unsigned short* __restrict__ k_t, unsigned short* __restrict__ q_t,
    unsigned short* __restrict__ v_out) {
  const int lane = threadIdx.x & 31;
  const int W = blockIdx.x * 8 + (threadIdx.x >> 5);   // 65536 waves
  const int tt = W & 15;
  const int lt = (W >> 4) & 63;
  const int g  = (W >> 10) & 7;
  const int n  = W >> 13;
  const int hi = lane >> 4;
  const int ln = lane & 15;
  const int lidx = lt * 16 + ln;

  const unsigned short* wsel;
  const float* bsel;
  int coBase, proj;
  if (tt < 4)      { proj = 0; wsel = kw; bsel = kb; coBase = g * 64 + tt * 16; }
  else if (tt < 8) { proj = 1; wsel = qw; bsel = qb; coBase = g * 64 + (tt - 4) * 16; }
  else             { proj = 2; wsel = vw; bsel = vb; coBase = g * 128 + (tt - 8) * 16; }

  const unsigned short* brow = tok_t + ((size_t)n * L_ + lidx) * CT_ + g * 128;
  const unsigned short* arow = wsel + (size_t)(coBase + ln) * 128;
  const int a0off = hi * 16, a1off = hi * 16 + 8;

  v8f c = {};
  #pragma unroll
  for (int ks = 0; ks < 4; ++ks) {
    v16bf b = load_b(brow + ks * 32 + hi * 16);
    v16bf a = load_a(arow + ks * 32, a0off, a1off);
    c = wmma_bf16(a, b, c);
  }

  const int coL = coBase + hi * 8;   // lane holds 8 consecutive cout rows (C layout m = r + 8*hi)
  float vals[8];
  #pragma unroll
  for (int r = 0; r < 8; ++r) vals[r] = c[r] + bsel[coL + r];

  if (proj < 2) {
    // transposed store: k_t/q_t [n,h,l,32]; d = coL&31 .. +7 contiguous -> one 16B store
    unsigned short* dst = (proj == 0) ? k_t : q_t;
    const int hh = coL >> 5;
    const int d  = coL & 31;
    store8bf(dst + (((size_t)n * H_ + hh) * L_ + lidx) * 32 + d, vals);
  } else {
    // v natural [n,cout,L] (rows contiguous in L = K-dim of second attention GEMM)
    #pragma unroll
    for (int r = 0; r < 8; ++r)
      v_out[((size_t)n * CT_ + coL + r) * L_ + lidx] = f2bf(vals[r]);
  }
}

// ---------------- flash attention, double-buffered async K/V staging
// Block = 8 waves sharing one (n,h); wave w owns q-tile qg*8+w.
// Waves 0-3 stage K(2KB)+V chunk, waves 4-7 stage V chunk only -> per-wave async
// counts are 2 / 1, so the split-wait leaves exactly the next block's prefetch in flight.
__global__ __launch_bounds__(256) void attn_kernel(
    const unsigned short* __restrict__ k_t,
    const unsigned short* __restrict__ q_t,
    const unsigned short* __restrict__ v_in,
    const float* __restrict__ tokens,
    float* __restrict__ t2,
    unsigned short* __restrict__ t2t) {
  __shared__ __align__(16) unsigned short ldsK[2][32 * 32];  // 2 x 2 KB
  __shared__ __align__(16) unsigned short ldsV[2][64 * 32];  // 2 x 4 KB

  const int tid  = threadIdx.x;
  const int lane = tid & 31;
  const int wv   = tid >> 5;
  const int b    = blockIdx.x;        // 1024 blocks: n(8) x h(16) x qg(8)
  const int qg   = b & 7;
  const int h    = (b >> 3) & 15;
  const int n    = b >> 7;
  const int hi = lane >> 4;
  const int ln = lane & 15;
  const int qidx = (qg * 8 + wv) * 16 + ln;

  const unsigned short* kbase = k_t + ((size_t)n * H_ + h) * (size_t)L_ * 32;
  const unsigned short* vbase = v_in + ((size_t)n * CT_ + h * 64) * (size_t)L_;

  // stage k-block kb into buffer buf
  auto stage = [&](int kb, int buf) {
    const int k0 = kb * 32;
    if (tid < 128)   // K block: rows k0..k0+31 are one contiguous 2KB chunk
      async_copy16((const char*)(kbase + (size_t)k0 * 32) + tid * 16,
                   (char*)ldsK[buf] + tid * 16);
    // V block: 64 rows of 64B (4 x 16B chunks per row)
    async_copy16((const char*)(vbase + (size_t)(tid >> 2) * L_ + k0) + (tid & 3) * 16,
                 (char*)ldsV[buf] + tid * 16);
  };

  // Q tile [d=32 x q=16] as B-operand, registers for all 32 k-blocks
  const v16bf bq =
      load_b(q_t + (((size_t)n * H_ + h) * L_ + qidx) * 32 + hi * 16);

  v8f acc[4];
  acc[0] = {}; acc[1] = {}; acc[2] = {}; acc[3] = {};
  float mRun = -3.0e38f, lRun = 0.0f;
  const int a0off = hi * 16, a1off = hi * 16 + 8;

  stage(0, 0);   // prologue

  for (int kb = 0; kb < 32; ++kb) {
    const int cur = kb & 1;
    const bool hasNext = (kb + 1) < 32;
    if (hasNext) stage(kb + 1, cur ^ 1);
    // wait for block kb's data; allow the just-issued prefetch to stay in flight
    if (hasNext) {
      if (tid < 128) { ASYNC_WAIT(2); } else { ASYNC_WAIT(1); }   // wave-uniform
    } else {
      ASYNC_WAIT(0);
    }
    __syncthreads();

    const unsigned short* bK = ldsK[cur];
    const unsigned short* bV = ldsV[cur];

    // S = K^T Q (1/32 scale folded into K projection)
    v16bf ak0 = load_a(bK + (size_t)ln * 32, a0off, a1off);
    v16bf ak1 = load_a(bK + (size_t)(16 + ln) * 32, a0off, a1off);
    v8f s0 = {}, s1 = {};
    s0 = wmma_bf16(ak0, bq, s0);
    s1 = wmma_bf16(ak1, bq, s1);

    // per-q-column running max/sum: lane l and l+16 hold the two M-halves of column l&15
    float bm = -3.0e38f;
    #pragma unroll
    for (int r = 0; r < 8; ++r) {
      bm = fmaxf(bm, s0[r]);
      bm = fmaxf(bm, s1[r]);
    }
    bm = fmaxf(bm, __shfl_xor(bm, 16, 32));
    const float mNew  = fmaxf(mRun, bm);
    const float alpha = __expf(mRun - mNew);

    float p0[8], p1[8], ps = 0.0f;
    #pragma unroll
    for (int r = 0; r < 8; ++r) {
      p0[r] = __expf(s0[r] - mNew);
      p1[r] = __expf(s1[r] - mNew);
      ps += p0[r] + p1[r];
    }
    ps += __shfl_xor(ps, 16, 32);
    lRun = lRun * alpha + ps;
    mRun = mNew;

    #pragma unroll
    for (int t = 0; t < 4; ++t)
      #pragma unroll
      for (int r = 0; r < 8; ++r)
        acc[t][r] *= alpha;

    // reshape P (two C-layout 16x16 tiles) into one 32x16 B-operand via lane-half swap
    BF16x16 bp;
    #pragma unroll
    for (int r = 0; r < 8; ++r) {
      const float x0 = __shfl_xor(p0[r], 16, 32);
      const float x1 = __shfl_xor(p1[r], 16, 32);
      bp.u[r]     = f2bf(hi ? x1 : p0[r]);     // K = hi*16 + r
      bp.u[8 + r] = f2bf(hi ? p1[r] : x0);     // K = hi*16 + 8 + r
    }

    // acc[d-tile] += V_tile(16x32) * P(32x16)
    #pragma unroll
    for (int t = 0; t < 4; ++t) {
      v16bf av = load_a(bV + (size_t)(t * 16 + ln) * 32, a0off, a1off);
      acc[t] = wmma_bf16(av, bp.v, acc[t]);
    }
    __syncthreads();   // buffer `cur` may be overwritten by stage(kb+2) next iteration
  }

  const float inv = 1.0f / lRun;
  #pragma unroll
  for (int t = 0; t < 4; ++t) {
    const int co0 = h * 64 + t * 16 + hi * 8;
    float vals[8];
    #pragma unroll
    for (int r = 0; r < 8; ++r) {
      const size_t idx = ((size_t)n * CT_ + co0 + r) * L_ + qidx;
      const float y = acc[t][r] * inv + tokens[idx];   // residual #1
      t2[idx] = y;                                     // fp32 for residual #2
      vals[r] = y;
    }
    // bf16 transposed copy [n,l,ci] for the ff GEMM B-operand (16B store)
    store8bf(t2t + ((size_t)n * L_ + qidx) * CT_ + co0, vals);
  }
}

// ---------------- ff conv (1024-K GEMM) + bias + residual
// Block = 8 waves sharing one cout-tile; 32KB weight strip staged to LDS once (async).
__global__ __launch_bounds__(256) void ff_kernel(
    const unsigned short* __restrict__ t2t,
    const float* __restrict__ t2,
    const unsigned short* __restrict__ fw,
    const float* __restrict__ fb,
    float* __restrict__ out) {
  __shared__ __align__(16) unsigned short ldsA[16 * CT_];   // 32 KB weight strip

  const int tid  = threadIdx.x;
  const int lane = tid & 31;
  const int wv   = tid >> 5;
  const int b    = blockIdx.x;        // 4096 blocks: cot(64) x colg(64)
  const int cot  = b >> 6;
  const int colg = b & 63;
  const int colt = colg * 8 + wv;
  const int n  = colt >> 6;
  const int lt = colt & 63;
  const int hi = lane >> 4;
  const int ln = lane & 15;
  const int lcol = lt * 16 + ln;

  // stage the 16 contiguous weight rows (cot*16 .. +15) = 32768B
  {
    const char* ag = (const char*)(fw + (size_t)cot * 16 * CT_);
    #pragma unroll
    for (int i = 0; i < 8; ++i) {
      const int c16 = tid + i * 256;
      async_copy16(ag + (size_t)c16 * 16, (char*)ldsA + (size_t)c16 * 16);
    }
    ASYNC_WAIT(0);
    __syncthreads();
  }

  const unsigned short* brow = t2t + ((size_t)n * L_ + lcol) * CT_;
  const unsigned short* arow = ldsA + (size_t)ln * CT_;
  const int a0off = hi * 16, a1off = hi * 16 + 8;

  v8f c = {};
  for (int ks = 0; ks < 32; ++ks) {
    v16bf bv = load_b(brow + ks * 32 + hi * 16);
    v16bf av = load_a(arow + ks * 32, a0off, a1off);
    c = wmma_bf16(av, bv, c);
  }

  const int co0 = cot * 16 + hi * 8;
  #pragma unroll
  for (int r = 0; r < 8; ++r) {
    const size_t idx = ((size_t)n * CT_ + co0 + r) * L_ + lcol;
    out[idx] = t2[idx] + c[r] + fb[co0 + r];
  }
}

extern "C" void kernel_launch(void* const* d_in, const int* in_sizes, int n_in,
                              void* d_out, int out_size, void* d_ws, size_t ws_size,
                              hipStream_t stream) {
  (void)in_sizes; (void)n_in; (void)out_size; (void)ws_size;

  const float* tokens = (const float*)d_in[0];
  const float* k_w = (const float*)d_in[1];
  const float* k_b = (const float*)d_in[2];
  const float* k_g = (const float*)d_in[3];
  const float* k_be= (const float*)d_in[4];
  const float* k_m = (const float*)d_in[5];
  const float* k_v = (const float*)d_in[6];
  const float* q_w = (const float*)d_in[7];
  const float* q_b = (const float*)d_in[8];
  const float* q_g = (const float*)d_in[9];
  const float* q_be= (const float*)d_in[10];
  const float* q_m = (const float*)d_in[11];
  const float* q_v = (const float*)d_in[12];
  const float* v_w = (const float*)d_in[13];
  const float* v_b = (const float*)d_in[14];
  const float* v_g = (const float*)d_in[15];
  const float* v_be= (const float*)d_in[16];
  const float* v_m = (const float*)d_in[17];
  const float* v_v = (const float*)d_in[18];
  const float* f_w = (const float*)d_in[19];
  const float* f_b = (const float*)d_in[20];
  const float* f_g = (const float*)d_in[21];
  const float* f_be= (const float*)d_in[22];
  const float* f_m = (const float*)d_in[23];
  const float* f_v = (const float*)d_in[24];

  char* ws = (char*)d_ws;
  size_t off = 0;
  auto take = [&](size_t bytes) -> char* {
    char* p = ws + off;
    off += (bytes + 255) & ~(size_t)255;
    return p;
  };
  unsigned short* tok_t = (unsigned short*)take((size_t)N_ * L_ * CT_ * 2);     // 16 MB
  unsigned short* k_t   = (unsigned short*)take((size_t)N_ * H_ * L_ * 32 * 2); // 8 MB
  unsigned short* q_t   = (unsigned short*)take((size_t)N_ * H_ * L_ * 32 * 2); // 8 MB
  unsigned short* v_bf  = (unsigned short*)take((size_t)N_ * CT_ * L_ * 2);     // 16 MB
  float*          t2    = (float*)take((size_t)N_ * CT_ * L_ * 4);              // 32 MB
  unsigned short* t2t   = (unsigned short*)take((size_t)N_ * L_ * CT_ * 2);     // 16 MB
  unsigned short* kwb   = (unsigned short*)take(512 * 128 * 2);
  unsigned short* qwb   = (unsigned short*)take(512 * 128 * 2);
  unsigned short* vwb   = (unsigned short*)take(1024 * 128 * 2);
  unsigned short* fwb   = (unsigned short*)take((size_t)1024 * 1024 * 2);
  float* kbb = (float*)take(512 * 4);
  float* qbb = (float*)take(512 * 4);
  float* vbb = (float*)take(1024 * 4);
  float* fbb = (float*)take(1024 * 4);

  // fold BN (and the 1/sqrt(L)=1/32 attention scale into the K projection)
  fold_bn_kernel<<<256, 256, 0, stream>>>(k_w, k_b, k_g, k_be, k_m, k_v, 512, 128,
                                          0.03125f, kwb, kbb);
  fold_bn_kernel<<<256, 256, 0, stream>>>(q_w, q_b, q_g, q_be, q_m, q_v, 512, 128,
                                          1.0f, qwb, qbb);
  fold_bn_kernel<<<512, 256, 0, stream>>>(v_w, v_b, v_g, v_be, v_m, v_v, 1024, 128,
                                          1.0f, vwb, vbb);
  fold_bn_kernel<<<1024, 256, 0, stream>>>(f_w, f_b, f_g, f_be, f_m, f_v, 1024, 1024,
                                           1.0f, fwb, fbb);

  transpose_kernel<<<dim3(32, 32, 8), dim3(32, 8), 0, stream>>>(tokens, tok_t);

  qkv_kernel<<<8192, 256, 0, stream>>>(tok_t, kwb, kbb, qwb, qbb, vwb, vbb,
                                       k_t, q_t, v_bf);

  attn_kernel<<<1024, 256, 0, stream>>>(k_t, q_t, v_bf, tokens, t2, t2t);

  ff_kernel<<<4096, 256, 0, stream>>>(t2t, t2, fwb, fbb, (float*)d_out);
}
